// DeformAtten1D_45784351375775
// MI455X (gfx1250) — compile-verified, matrix operand order checked by reading, not run on hardware
//
#include <hip/hip_runtime.h>
#include <hip/hip_bf16.h>
#include <math.h>

// Problem sizes (fixed by the reference)
#define BB 4
#define SS 4096
#define DD 1024
#define HH 16
#define GG 4
#define KK 3
#define DG 256   // D / G
#define DH 64    // D / H
#define SCALE (1.0f / 32.0f)   // D^-0.5

typedef __attribute__((ext_vector_type(16))) __bf16 bf16x16;
typedef __attribute__((ext_vector_type(8)))  __bf16 bf16x8;
typedef __attribute__((ext_vector_type(4)))  __bf16 bf16x4;
typedef __attribute__((ext_vector_type(8)))  float  f32x8;
typedef __attribute__((ext_vector_type(4)))  unsigned int u32x4;
typedef __attribute__((ext_vector_type(8)))  int i32x8;
typedef __attribute__((ext_vector_type(4)))  int i32x4;

// LDS staging layout for the 64x1024 bf16 Wq tile: data row = 2048 B; the TDM
// pad feature inserts 2 dwords (8 B) after every 256 dwords (1024 B) of data,
// giving an LDS row stride of 2064 B = 516 dwords -> lane-to-lane bank stride
// of 4 dwords -> conflict-free ds_load_b128 B-fragment reads.
#define WTILE_N     64
#define WROW_BYTES  2048
#define LDS_BYTES   (WTILE_N * (WROW_BYTES + 16))   // 132096
__device__ __forceinline__ unsigned lds_off(unsigned dataoff) {
    return dataoff + ((dataoff >> 10) << 3);        // +8B per 1KB of data
}

// ---------------------------------------------------------------------------
// K0: f32 -> bf16 elementwise convert (x and Wq), vectorized x4
// ---------------------------------------------------------------------------
__global__ void k_convert(const float* __restrict__ src, __bf16* __restrict__ dst, int n4) {
    int i = blockIdx.x * blockDim.x + threadIdx.x;
    if (i >= n4) return;
    float4 v = ((const float4*)src)[i];
    bf16x4 o;
    o.x = (__bf16)v.x; o.y = (__bf16)v.y; o.z = (__bf16)v.z; o.w = (__bf16)v.w;
    ((bf16x4*)dst)[i] = o;
}

// ---------------------------------------------------------------------------
// K1: collapse the offset head: Weff[c,dk] = sum_o Woff2[o]*Woff1[o,c,dk],
//     weff[DG*KK] = beff = dot(Woff2, boff1).  One block of 256 threads.
// ---------------------------------------------------------------------------
__global__ void k_weff(const float* __restrict__ Woff1, const float* __restrict__ boff1,
                       const float* __restrict__ Woff2, float* __restrict__ weff) {
    int c = threadIdx.x;           // 0..255
    float acc[KK] = {0.f, 0.f, 0.f};
    for (int o = 0; o < DG; ++o) {
        float w2 = Woff2[o];
        const float* p = Woff1 + (size_t)(o * DG + c) * KK;
        #pragma unroll
        for (int dk = 0; dk < KK; ++dk) acc[dk] += w2 * p[dk];
    }
    #pragma unroll
    for (int dk = 0; dk < KK; ++dk) weff[c * KK + dk] = acc[dk];
    if (c == 0) {
        float be = 0.f;
        for (int o = 0; o < DG; ++o) be += Woff2[o] * boff1[o];
        weff[DG * KK] = be;
    }
}

// ---------------------------------------------------------------------------
// K2: Q projection GEMM, WMMA bf16 -> f32 accum, TDM-staged B tile in LDS.
// Q[b,s,o] = sum_c X[b,s,c]*Wq[o,c] + bq[o], stored bf16 [B,S,D].
// Block: 256 threads = 8 waves; block tile 256(M) x 64(N); wave tile 32 x 64.
// The block's Wq[n0..n0+63, :] tile (128 KB bf16) is DMA'd into LDS once by
// the Tensor Data Mover and shared by all 8 waves (8x less L2 traffic on B).
// A fragment (16x32 bf16): lane l -> row (l&15), K = hb*8 + {0..7, 16..23};
// B fragment (32x16 bf16): lane l -> col (l&15), same K pattern (per ISA
// layouts) -> two 16-byte reads per lane, global (A) / LDS (B).
// ---------------------------------------------------------------------------
__global__ void __launch_bounds__(256) k_qgemm(const __bf16* __restrict__ X,
                                               const __bf16* __restrict__ W,
                                               const float*  __restrict__ bq,
                                               __bf16* __restrict__ Q) {
    extern __shared__ char lsB[];        // padded 64x1024 bf16 W tile (132096 B)

    const int wave = threadIdx.x >> 5;
    const int lane = threadIdx.x & 31;
    const int ln   = lane & 15;
    const int hb   = lane >> 4;          // 0 or 1 -> K sub-offset of 8
    const int b    = blockIdx.z;
    const int m0   = blockIdx.x * 256 + wave * 32;
    const int n0   = blockIdx.y * 64;

    // ---- stage Wq tile -> LDS --------------------------------------------
#if __has_builtin(__builtin_amdgcn_tensor_load_to_lds) && __has_builtin(__builtin_amdgcn_s_wait_tensorcnt)
    if (threadIdx.x < 32) {
        // D# group0: count=1 | lds_addr=0 (only LDS alloc) | global_addr | type=2
        unsigned long long ga = (unsigned long long)(const void*)(W + (size_t)n0 * DD);
        u32x4 g0;
        g0[0] = 1u;
        g0[1] = 0u;
        g0[2] = (unsigned)(ga & 0xffffffffull);
        g0[3] = (unsigned)((ga >> 32) & 0x01ffffffull) | 0x80000000u;  // type=2 ("image")
        // D# group1: data_size=2B, pad_enable, pad_interval=256dw, pad_amount=2dw,
        // tensor_dim0=1024, tensor_dim1=64, tile_dim0=1024, tile_dim1=64,
        // tensor_dim0_stride=1024 (elements).
        i32x8 g1;
        g1[0] = (int)0x03D10000;        // dsz=1<<16 | pad_en<<20 | 7<<22 | 1<<25
        g1[1] = (int)(1024u << 16);     // tensor_dim0[15:0] in bits[63:48]
        g1[2] = (int)(64u << 16);       // tensor_dim0 msbs=0 | tensor_dim1[15:0]
        g1[3] = (int)(1024u << 16);     // tensor_dim1 msbs=0 | tile_dim0=1024
        g1[4] = 64;                     // tile_dim1=64, tile_dim2=0
        g1[5] = 1024;                   // tensor_dim0_stride lo
        g1[6] = 0;
        g1[7] = 0;
        i32x4 z4 = {0, 0, 0, 0};
  #if __clang_major__ >= 23
        i32x8 z8 = {0, 0, 0, 0, 0, 0, 0, 0};
        __builtin_amdgcn_tensor_load_to_lds(g0, g1, z4, z4, z8, 0);
  #else
        __builtin_amdgcn_tensor_load_to_lds(g0, g1, z4, z4, 0);
  #endif
        __builtin_amdgcn_s_wait_tensorcnt(0);
    }
#else
    // Fallback: cooperative copy into the same padded layout.
    {
        const char* wsrc = (const char*)(W + (size_t)n0 * DD);
        for (int chunk = threadIdx.x; chunk < (WTILE_N * WROW_BYTES) / 16; chunk += 256) {
            unsigned doff = chunk * 16;
            *(bf16x8*)(lsB + lds_off(doff)) = *(const bf16x8*)(wsrc + doff);
        }
    }
#endif
    __syncthreads();

    // ---- GEMM main loop ---------------------------------------------------
    const __bf16* xrow0 = X + ((size_t)(b * SS) + (m0 + ln)) * DD;        // A rows
    const __bf16* xrow1 = xrow0 + (size_t)16 * DD;
    f32x8 acc[8] = {};

    for (int k0 = 0; k0 < DD; k0 += 32) {
        __builtin_prefetch(xrow0 + k0 + 128, 0, 1);   // global_prefetch_b8
        bf16x16 A0, A1;
        {
            bf16x8 lo = *(const bf16x8*)(xrow0 + k0 + hb * 8);
            bf16x8 hi = *(const bf16x8*)(xrow0 + k0 + 16 + hb * 8);
            #pragma unroll
            for (int i = 0; i < 8; ++i) { A0[i] = lo[i]; A0[8 + i] = hi[i]; }
        }
        {
            bf16x8 lo = *(const bf16x8*)(xrow1 + k0 + hb * 8);
            bf16x8 hi = *(const bf16x8*)(xrow1 + k0 + 16 + hb * 8);
            #pragma unroll
            for (int i = 0; i < 8; ++i) { A1[i] = lo[i]; A1[8 + i] = hi[i]; }
        }
        #pragma unroll
        for (int f = 0; f < 4; ++f) {
            unsigned row = (unsigned)(f * 16 + ln);
            unsigned d0 = row * WROW_BYTES + (unsigned)(k0 + hb * 8) * 2;
            unsigned d1 = row * WROW_BYTES + (unsigned)(k0 + 16 + hb * 8) * 2;
            bf16x8 blo = *(const bf16x8*)(lsB + lds_off(d0));   // ds_load_b128
            bf16x8 bhi = *(const bf16x8*)(lsB + lds_off(d1));
            bf16x16 Bf;
            #pragma unroll
            for (int i = 0; i < 8; ++i) { Bf[i] = blo[i]; Bf[8 + i] = bhi[i]; }
            acc[f] = __builtin_amdgcn_wmma_f32_16x16x32_bf16(
                false, A0, false, Bf, (short)0, acc[f], false, false);
            acc[4 + f] = __builtin_amdgcn_wmma_f32_16x16x32_bf16(
                false, A1, false, Bf, (short)0, acc[4 + f], false, false);
        }
    }

    // C/D layout: VGPR r -> M = mbase + r + hb*8 ; lane -> N = n0 + f*16 + ln
    #pragma unroll
    for (int f = 0; f < 4; ++f) {
        int n = n0 + f * 16 + ln;
        float bias = bq[n];
        #pragma unroll
        for (int r = 0; r < 8; ++r) {
            int ma = m0 + r + hb * 8;
            int mb = ma + 16;
            Q[((size_t)(b * SS) + ma) * DD + n] = (__bf16)(acc[f][r] + bias);
            Q[((size_t)(b * SS) + mb) * DD + n] = (__bf16)(acc[4 + f][r] + bias);
        }
    }
}

// ---------------------------------------------------------------------------
// K3: offset -> wx.  One wave per (b,g,s).  Lane l covers channels l*8..l*8+7
// across the 3 taps; cross-lane reduce via shuffles; lane 0 applies tanh and
// the grid-sample bilinear width weight:
//   wx = clamp(1 - |(s+offset)/(S-1) - 0.5|, 0, 1)
// ---------------------------------------------------------------------------
__global__ void __launch_bounds__(256) k_offset(const __bf16* __restrict__ Q,
                                                const float* __restrict__ weff,
                                                float* __restrict__ wx) {
    int wglobal = blockIdx.x * 8 + (threadIdx.x >> 5);
    int lane = threadIdx.x & 31;
    int s = wglobal % SS;
    int g = (wglobal / SS) % GG;
    int b = wglobal / (SS * GG);

    const __bf16* qbase = Q + (size_t)(b * SS) * DD + g * DG + lane * 8;
    float acc = 0.f;
    #pragma unroll
    for (int dk = 0; dk < KK; ++dk) {
        int ss = s + dk - 1;
        if (ss < 0 || ss >= SS) continue;   // zero padding of the conv
        bf16x8 qv = *(const bf16x8*)(qbase + (size_t)ss * DD);
        #pragma unroll
        for (int i = 0; i < 8; ++i)
            acc += (float)qv[i] * weff[(lane * 8 + i) * KK + dk];
    }
    #pragma unroll
    for (int off = 16; off > 0; off >>= 1) acc += __shfl_xor(acc, off, 32);

    if (lane == 0) {
        float offset = tanhf(acc + weff[DG * KK]) * (float)KK;
        float ix = ((float)s + offset) * (1.0f / (float)(SS - 1)) - 0.5f;
        float w = 1.0f - fabsf(ix);
        wx[(b * GG + g) * SS + s] = fmaxf(0.f, fminf(1.f, w));
    }
}

// ---------------------------------------------------------------------------
// K4: feat_c + rank-G factors of K and V:
//   Kg[b,o,g] = sum_c Wk[o, g*DG+c] * 0.5*(x[b,2047,g*DG+c] + x[b,2048,g*DG+c])
// (iy = (S-1)/2 is constant -> wy0 = wy1 = 0.5).  Thread per (b,o,g).
// ---------------------------------------------------------------------------
__global__ void k_kgvg(const float* __restrict__ x, const float* __restrict__ Wk,
                       const float* __restrict__ Wv, float* __restrict__ kg,
                       float* __restrict__ vg) {
    int idx = blockIdx.x * blockDim.x + threadIdx.x;
    if (idx >= BB * DD * GG) return;
    int g = idx % GG;
    int o = (idx / GG) % DD;
    int b = idx / (GG * DD);
    const float* xr0 = x + ((size_t)(b * SS) + (SS / 2 - 1)) * DD + g * DG;  // s = 2047
    const float* xr1 = xr0 + DD;                                            // s = 2048
    const float* wkr = Wk + (size_t)o * DD + g * DG;
    const float* wvr = Wv + (size_t)o * DD + g * DG;
    float ak = 0.f, av_ = 0.f;
    for (int c = 0; c < DG; ++c) {
        float f = 0.5f * (xr0[c] + xr1[c]);
        ak += wkr[c] * f;
        av_ += wvr[c] * f;
    }
    kg[((size_t)(b * DD) + o) * GG + g] = ak;
    vg[((size_t)(b * DD) + o) * GG + g] = av_;
}

// K5a: zero the atomic accumulators (qw: B*D*G, qs: B*D floats, contiguous)
__global__ void k_zero(float* __restrict__ p, int n) {
    int i = blockIdx.x * blockDim.x + threadIdx.x;
    if (i < n) p[i] = 0.f;
}

// ---------------------------------------------------------------------------
// K5: Qw[b,o,g] = sum_s Q[b,s,o]*wx[b,g,s];  Qs[b,o] = sum_s Q[b,s,o].
// Block covers 256 consecutive o for one (b, s-chunk of 256); atomics merge.
// ---------------------------------------------------------------------------
__global__ void __launch_bounds__(256) k_qw(const __bf16* __restrict__ Q,
                                            const float* __restrict__ wx,
                                            float* __restrict__ qw,
                                            float* __restrict__ qs) {
    int o = blockIdx.x * 256 + threadIdx.x;
    int b = blockIdx.z;
    int s0 = blockIdx.y * 256;
    float aqs = 0.f, aqw[GG] = {0.f, 0.f, 0.f, 0.f};
    for (int i = 0; i < 256; ++i) {
        int s = s0 + i;
        float q = (float)Q[((size_t)(b * SS) + s) * DD + o];
        aqs += q;
        #pragma unroll
        for (int g = 0; g < GG; ++g) aqw[g] += q * wx[(b * GG + g) * SS + s];
    }
    #pragma unroll
    for (int g = 0; g < GG; ++g)
        atomicAdd(&qw[((size_t)(b * DD) + o) * GG + g], aqw[g]);
    atomicAdd(&qs[b * DD + o], aqs);
}

// ---------------------------------------------------------------------------
// K6: per-(b,h) 64x64 attention in the collapsed basis:
//   scores[i,j] = scale*(sum_g Qw[i,g]*Kg[j,g] + Qs[i]*bk[j])
//   attn = softmax_j;  AV[i,g] = attn@Vg;  ab[i] = attn@bv
// ---------------------------------------------------------------------------
__global__ void __launch_bounds__(64) k_attn(const float* __restrict__ qw,
                                             const float* __restrict__ qs,
                                             const float* __restrict__ kg,
                                             const float* __restrict__ vg,
                                             const float* __restrict__ bk,
                                             const float* __restrict__ bv,
                                             float* __restrict__ av,
                                             float* __restrict__ ab) {
    int b = blockIdx.x / HH;
    int h = blockIdx.x % HH;
    int i = threadIdx.x;
    int ip = h * DH + i;
    float qwr[GG];
    #pragma unroll
    for (int g = 0; g < GG; ++g) qwr[g] = qw[((size_t)(b * DD) + ip) * GG + g];
    float qsi = qs[b * DD + ip];

    float sc[DH];
    float mx = -1e30f;
    for (int j = 0; j < DH; ++j) {
        int jp = h * DH + j;
        float v = 0.f;
        #pragma unroll
        for (int g = 0; g < GG; ++g) v += qwr[g] * kg[((size_t)(b * DD) + jp) * GG + g];
        v = (v + qsi * bk[jp]) * SCALE;
        sc[j] = v;
        mx = fmaxf(mx, v);
    }
    float sum = 0.f;
    for (int j = 0; j < DH; ++j) { float e = __expf(sc[j] - mx); sc[j] = e; sum += e; }
    float inv = 1.0f / sum;

    #pragma unroll
    for (int g = 0; g < GG; ++g) {
        float a = 0.f;
        for (int j = 0; j < DH; ++j)
            a += sc[j] * vg[((size_t)(b * DD) + h * DH + j) * GG + g];
        av[((size_t)(b * DD) + ip) * GG + g] = a * inv;
    }
    float abv = 0.f;
    for (int j = 0; j < DH; ++j) abv += sc[j] * bv[h * DH + j];
    ab[b * DD + ip] = abv * inv;
}

// ---------------------------------------------------------------------------
// K7: fold the output projection into the rank-G basis:
//   WAV[b,o,g] = sum_i Wo[o,i]*AV[b,i,g];  c1[b,o] = sum_i Wo[o,i]*ab[b,i] + bo[o]
//   rowsum[o]  = sum_i Wo[o,i]   (softmax rows sum to 1 -> bias_table term)
// ---------------------------------------------------------------------------
__global__ void k_wav(const float* __restrict__ Wo, const float* __restrict__ bo,
                      const float* __restrict__ av, const float* __restrict__ ab,
                      float* __restrict__ wav, float* __restrict__ c1,
                      float* __restrict__ rowsum) {
    int idx = blockIdx.x * blockDim.x + threadIdx.x;
    if (idx >= BB * DD) return;
    int o = idx % DD;
    int b = idx / DD;
    float w4[GG] = {0.f, 0.f, 0.f, 0.f};
    float c = 0.f, rs = 0.f;
    const float* wrow = Wo + (size_t)o * DD;
    for (int ip = 0; ip < DD; ++ip) {
        float w = wrow[ip];
        #pragma unroll
        for (int g = 0; g < GG; ++g) w4[g] += w * av[((size_t)(b * DD) + ip) * GG + g];
        c += w * ab[b * DD + ip];
        rs += w;
    }
    #pragma unroll
    for (int g = 0; g < GG; ++g) wav[((size_t)(b * DD) + o) * GG + g] = w4[g];
    c1[b * DD + o] = c + bo[o];
    if (b == 0) rowsum[o] = rs;
}

// ---------------------------------------------------------------------------
// K8: expand the rank-G output (write-bandwidth bound, 64 MB):
//   y[b,s,o] = sum_g wx[b,g,s]*WAV[b,o,g] + c1[b,o] + bias_table[s]*rowsum[o]
// ---------------------------------------------------------------------------
__global__ void __launch_bounds__(256) k_final(const float* __restrict__ wx,
                                               const float* __restrict__ wav,
                                               const float* __restrict__ c1,
                                               const float* __restrict__ rowsum,
                                               const float* __restrict__ bias_table,
                                               float* __restrict__ y) {
    int s = blockIdx.x;
    int b = blockIdx.y;
    float wxv[GG];
    #pragma unroll
    for (int g = 0; g < GG; ++g) wxv[g] = wx[(b * GG + g) * SS + s];
    float bt = bias_table[s];
    #pragma unroll
    for (int oc = 0; oc < 4; ++oc) {
        int o = oc * 256 + threadIdx.x;
        const float* wv = &wav[((size_t)(b * DD) + o) * GG];
        float r = c1[b * DD + o] + bt * rowsum[o];
        #pragma unroll
        for (int g = 0; g < GG; ++g) r += wxv[g] * wv[g];
        y[((size_t)(b * SS) + s) * DD + o] = r;
    }
}

// ---------------------------------------------------------------------------
extern "C" void kernel_launch(void* const* d_in, const int* in_sizes, int n_in,
                              void* d_out, int out_size, void* d_ws, size_t ws_size,
                              hipStream_t stream) {
    const float* x     = (const float*)d_in[0];
    const float* Wq    = (const float*)d_in[1];
    const float* bq    = (const float*)d_in[2];
    const float* Wk    = (const float*)d_in[3];
    const float* bk    = (const float*)d_in[4];
    const float* Wv    = (const float*)d_in[5];
    const float* bv    = (const float*)d_in[6];
    const float* Wo    = (const float*)d_in[7];
    const float* bo    = (const float*)d_in[8];
    const float* Woff1 = (const float*)d_in[9];
    const float* boff1 = (const float*)d_in[10];
    const float* Woff2 = (const float*)d_in[11];
    const float* btab  = (const float*)d_in[12];
    float* y = (float*)d_out;

    // carve workspace
    char* w = (char*)d_ws;
    auto alloc = [&](size_t bytes) -> void* {
        void* p = (void*)w;
        w += (bytes + 255) & ~(size_t)255;
        return p;
    };
    __bf16* xbf   = (__bf16*)alloc((size_t)BB * SS * DD * 2);
    __bf16* wqbf  = (__bf16*)alloc((size_t)DD * DD * 2);
    __bf16* qbf   = (__bf16*)alloc((size_t)BB * SS * DD * 2);
    float*  wx    = (float*)alloc((size_t)BB * GG * SS * 4);
    float*  weff  = (float*)alloc((DG * KK + 1) * 4);
    float*  kg    = (float*)alloc((size_t)BB * DD * GG * 4);
    float*  vg    = (float*)alloc((size_t)BB * DD * GG * 4);
    float*  qw    = (float*)alloc((size_t)BB * DD * GG * 4);   // qw then qs contiguous
    float*  qs    = (float*)alloc((size_t)BB * DD * 4);
    float*  av    = (float*)alloc((size_t)BB * DD * GG * 4);
    float*  ab    = (float*)alloc((size_t)BB * DD * 4);
    float*  wav   = (float*)alloc((size_t)BB * DD * GG * 4);
    float*  c1    = (float*)alloc((size_t)BB * DD * 4);
    float*  rsum  = (float*)alloc((size_t)DD * 4);

    // K0: bf16 packs
    {
        int n4 = BB * SS * DD / 4;
        k_convert<<<n4 / 256, 256, 0, stream>>>(x, xbf, n4);
        int w4 = DD * DD / 4;
        k_convert<<<w4 / 256, 256, 0, stream>>>(Wq, wqbf, w4);
    }
    // K1: collapsed offset-head weights
    k_weff<<<1, DG, 0, stream>>>(Woff1, boff1, Woff2, weff);

    // K2: Q = X * Wq^T + bq (WMMA bf16, TDM-staged B tile)
    k_qgemm<<<dim3(SS / 256, DD / 64, BB), 256, LDS_BYTES, stream>>>(xbf, wqbf, bq, qbf);

    // K3: offsets -> wx
    k_offset<<<(BB * GG * SS) / 8, 256, 0, stream>>>(qbf, weff, wx);

    // K4: rank-G K/V factors
    k_kgvg<<<(BB * DD * GG) / 256, 256, 0, stream>>>(x, Wk, Wv, kg, vg);

    // K5: Qw / Qs (zero then atomic-accumulate)
    k_zero<<<(BB * DD * GG + BB * DD + 255) / 256, 256, 0, stream>>>(qw, BB * DD * GG + BB * DD);
    k_qw<<<dim3(DD / 256, SS / 256, BB), 256, 0, stream>>>(qbf, wx, qw, qs);

    // K6: attention in collapsed basis
    k_attn<<<BB * HH, DH, 0, stream>>>(qw, qs, kg, vg, bk, bv, av, ab);

    // K7: fold Wo
    k_wav<<<(BB * DD) / 256, 256, 0, stream>>>(Wo, bo, av, ab, wav, c1, rsum);

    // K8: expand output
    k_final<<<dim3(SS, BB), 256, 0, stream>>>(wx, wav, c1, rsum, btab, y);
}
